// LinkPredictionModel_730144441187
// MI455X (gfx1250) — matrix operand order repaired and checked by reading, not compiled, not run on hardware
//
#include <hip/hip_runtime.h>

#define CH 128  // IN_CH == HID_CH == 128

typedef __attribute__((ext_vector_type(16))) __bf16          v16bf;
typedef __attribute__((ext_vector_type(8)))  float           v8f;
typedef __attribute__((ext_vector_type(8)))  unsigned short  us8;

union Frag {
  v16bf vec;
  us8   h[2];
};

__device__ __forceinline__ unsigned short f2bf(float f) {
  // round-to-nearest-even f32 -> bf16
  union { float f; unsigned int u; } c; c.f = f;
  unsigned int u = c.u;
  return (unsigned short)((u + 0x7FFFu + ((u >> 16) & 1u)) >> 16);
}

__global__ void zero_f32(float* __restrict__ p, long long n) {
  long long i = (long long)blockIdx.x * blockDim.x + threadIdx.x;
  if (i < n) p[i] = 0.0f;
}

__global__ void degree_count(const long long* __restrict__ dst, float* __restrict__ deg, int E) {
  int e = blockIdx.x * blockDim.x + threadIdx.x;
  if (e < E) atomicAdd(&deg[dst[e]], 1.0f);
}

__global__ void degree_finalize(float* __restrict__ dinv, float* __restrict__ invdeg, int N) {
  int i = blockIdx.x * blockDim.x + threadIdx.x;
  if (i < N) {
    float d = dinv[i] + 1.0f;       // +1 self-loop
    dinv[i]   = rsqrtf(d);          // deg^{-1/2}
    invdeg[i] = 1.0f / d;           // deg^{-1} (self-loop norm)
  }
}

// One-time weight conversion: Wb[n*128 + k] = bf16(W[k*128 + n])
// (transposed so a lane's B-fragment = two runs of 8 contiguous K values of its column)
__global__ void convert_w_bf16(const float* __restrict__ W, unsigned short* __restrict__ Wb) {
  int i = blockIdx.x * blockDim.x + threadIdx.x;
  if (i < CH * CH) {
    int n = i >> 7, k = i & 127;
    Wb[i] = f2bf(W[k * CH + n]);
  }
}

// H[N,128] = A[N,128] @ W[128,128], bf16 WMMA with f32 accumulate.
// 256 threads = 8 waves; block stages a 16x128 A tile in LDS as bf16 once,
// wave w owns output cols [16w, 16w+16). Requires N % 16 == 0 (N=100000 -> 6250 blocks).
__global__ void gemm_wmma_bf16(const float* __restrict__ A,
                               const unsigned short* __restrict__ Wb,  // [128n][128k] bf16
                               float* __restrict__ H) {
  __shared__ unsigned short As[16 * CH];  // 16 rows x 128 K, bf16, row stride 128

  const int tid = threadIdx.x;
  const long long rbase = (long long)blockIdx.x * 16;

  // cooperative stage+convert: 2048 bf16 elements, 4 per thread (float4 in, b64 out)
#pragma unroll
  for (int j = 0; j < 2; ++j) {
    int i4 = (tid + 256 * j) * 4;          // element index, multiple of 4
    int r = i4 >> 7, k = i4 & 127;
    float4 xf = *(const float4*)(A + (rbase + r) * CH + k);
    unsigned long long p = (unsigned long long)f2bf(xf.x)
                         | ((unsigned long long)f2bf(xf.y) << 16)
                         | ((unsigned long long)f2bf(xf.z) << 32)
                         | ((unsigned long long)f2bf(xf.w) << 48);
    *(unsigned long long*)(&As[i4]) = p;
  }
  __syncthreads();

  const int lane = tid & 31;
  const int wave = tid >> 5;
  const int half = lane >> 4;        // 0: lanes 0-15, 1: lanes 16-31
  const int mn   = lane & 15;
  const int ncol = wave * 16 + mn;   // output column owned by this lane (B/C/D lane = N)

  const unsigned short* wcol = Wb + ncol * CH;

  v8f c = {};
#pragma unroll
  for (int kb = 0; kb < CH; kb += 32) {
    Frag a, b;
    // A 16x32 bf16 layout: lane row = mn; VGPR0-3: K = kb+8*half+0..7, VGPR4-7: +16
    const unsigned short* ar = &As[mn * CH + kb + 8 * half];
    a.h[0] = *(const us8*)(ar);        // ds_load_b128
    a.h[1] = *(const us8*)(ar + 16);   // ds_load_b128
    // B 32x16 bf16 layout: lane col = mn(+wave*16); VGPR0-3: K = kb+8*half+0..7, VGPR4-7: +16
    const unsigned short* br = wcol + kb + 8 * half;
    b.h[0] = *(const us8*)(br);        // global_load_b128
    b.h[1] = *(const us8*)(br + 16);   // global_load_b128
    c = __builtin_amdgcn_wmma_f32_16x16x32_bf16(false, a.vec, false, b.vec,
                                                (short)0, c, false, false);
  }
  // C/D layout: VGPR r -> M = r + 8*half, lane -> N = mn
#pragma unroll
  for (int r = 0; r < 8; ++r) {
    long long mrow = rbase + r + 8 * half;
    H[mrow * CH + ncol] = c[r];
  }
}

// One wave per edge (grid-strided): lane l owns channels [4l, 4l+4).
// agg[dst] += h[src] * dinv[src] * dinv[dst]  via f32 atomics (resolve in 192MB L2).
__global__ void edge_scatter(const float* __restrict__ h,
                             const float* __restrict__ dinv,
                             const long long* __restrict__ src,
                             const long long* __restrict__ dst,
                             float* __restrict__ agg, int E) {
  const int lane = threadIdx.x & 31;
  const int nwaves = (gridDim.x * blockDim.x) >> 5;
  int w = (blockIdx.x * blockDim.x + threadIdx.x) >> 5;
  for (int e = w; e < E; e += nwaves) {
    long long s = src[e];
    long long d = dst[e];
    // prefetch next edge's gather row to hide random-access latency
    int en = e + nwaves;
    if (en < E) {
      long long sn = src[en];
      __builtin_prefetch(h + sn * CH + lane * 4, 0, 0);
    }
    float norm = dinv[s] * dinv[d];
    const float4 hv = *(const float4*)(h + s * CH + lane * 4);
    float* ap = agg + d * CH + lane * 4;
    atomicAdd(ap + 0, hv.x * norm);
    atomicAdd(ap + 1, hv.y * norm);
    atomicAdd(ap + 2, hv.z * norm);
    atomicAdd(ap + 3, hv.w * norm);
  }
}

// out = agg + h * invdeg[node] + bias[ch]; optional ReLU
__global__ void node_finalize(const float* __restrict__ agg,
                              const float* __restrict__ h,
                              const float* __restrict__ invdeg,
                              const float* __restrict__ bias,
                              float* __restrict__ out,
                              long long total, int relu) {
  long long i = (long long)blockIdx.x * blockDim.x + threadIdx.x;
  if (i < total) {
    long long node = i >> 7;  // /128
    int ch = (int)(i & (CH - 1));
    float v = agg[i] + h[i] * invdeg[node] + bias[ch];
    out[i] = relu ? fmaxf(v, 0.0f) : v;
  }
}

extern "C" void kernel_launch(void* const* d_in, const int* in_sizes, int n_in,
                              void* d_out, int out_size, void* d_ws, size_t ws_size,
                              hipStream_t stream) {
  const float*     x  = (const float*)d_in[0];
  const long long* ei = (const long long*)d_in[1];   // int64 [2, E]
  const float*     W1 = (const float*)d_in[2];
  const float*     b1 = (const float*)d_in[3];
  const float*     W2 = (const float*)d_in[4];
  const float*     b2 = (const float*)d_in[5];
  float* out = (float*)d_out;

  const long long N = in_sizes[0] / CH;   // 100000
  const int       E = in_sizes[1] / 2;    // 1600000
  const long long* src = ei;
  const long long* dst = ei + E;

  float* ws     = (float*)d_ws;
  float* dinv   = ws;                 // [N]   deg^{-1/2} (also deg accumulator)
  float* invdeg = ws + N;             // [N]   deg^{-1}
  float* h      = ws + 2 * N;         // [N*CH] linear output of current layer
  float* agg    = h + N * CH;         // [N*CH] scatter accumulator
  unsigned short* Wb1 = (unsigned short*)(agg + N * CH);  // [128*128] bf16, transposed
  unsigned short* Wb2 = Wb1 + CH * CH;

  const long long NC = N * CH;
  const int tpb = 256;
  const int nblkN  = (int)((N + tpb - 1) / tpb);
  const int nblkE  = (E + tpb - 1) / tpb;
  const int nblkNC = (int)((NC + tpb - 1) / tpb);
  const int nblkW  = (CH * CH + tpb - 1) / tpb;
  const int gemm_blocks = (int)(N / 16);  // N divisible by 16

  // --- one-time prep: degrees (shared by both layers) + bf16 weights ---
  zero_f32<<<nblkN, tpb, 0, stream>>>(dinv, N);
  degree_count<<<nblkE, tpb, 0, stream>>>(dst, dinv, E);
  degree_finalize<<<nblkN, tpb, 0, stream>>>(dinv, invdeg, (int)N);
  convert_w_bf16<<<nblkW, tpb, 0, stream>>>(W1, Wb1);
  convert_w_bf16<<<nblkW, tpb, 0, stream>>>(W2, Wb2);

  // --- layer 1: h = x@W1 ; agg = scatter ; out(tmp y1) = relu(agg + h/deg + b1) ---
  gemm_wmma_bf16<<<gemm_blocks, tpb, 0, stream>>>(x, Wb1, h);
  zero_f32<<<nblkNC, tpb, 0, stream>>>(agg, NC);
  edge_scatter<<<2048, tpb, 0, stream>>>(h, dinv, src, dst, agg, E);
  node_finalize<<<nblkNC, tpb, 0, stream>>>(agg, h, invdeg, b1, out, NC, 1);

  // --- layer 2: h = y1@W2 ; agg = scatter ; out = agg + h/deg + b2 ---
  gemm_wmma_bf16<<<gemm_blocks, tpb, 0, stream>>>(out, Wb2, h);
  zero_f32<<<nblkNC, tpb, 0, stream>>>(agg, NC);
  edge_scatter<<<2048, tpb, 0, stream>>>(h, dinv, src, dst, agg, E);
  node_finalize<<<nblkNC, tpb, 0, stream>>>(agg, h, invdeg, b2, out, NC, 0);
}
